// SELF_Attention_29523605193267
// MI455X (gfx1250) — compile-verified, matrix operand order checked by reading, not run on hardware
//
#include <hip/hip_runtime.h>

// ---------------------------------------------------------------------------
// SELF-Attention-ish reference on MI455X (gfx1250), bf16 WMMA pipeline.
//   K1: cvt h_s, W1, W2 -> bf16
//   K2: s1 = sig(h W1^T)*m, s2 = sig(h W2^T)*m  -> s1w(bf16), s1T(bf16),
//       s2(bf16), s2(f32)            [WMMA bf16]
//   K3: scores = (s1*w3) s2^T, mask, clip, softmax fused in LDS
//       -> probs f32 [b,i,j] + probs bf16 transposed pT [b,j,i]   [WMMA bf16]
//   K4: attn = s2f * (P^T s1)        [WMMA bf16, 32x64 per wave: every B tile
//       feeds two WMMAs -> 1.5 b128 loads per WMMA]
//   K5: ragged pairwise loss, deterministic tree reduction
// ---------------------------------------------------------------------------

static constexpr int kB = 16;
static constexpr int kL = 1024;
static constexpr int kD = 256;
static constexpr int kP = 2000;

typedef __attribute__((ext_vector_type(16))) __bf16 bf16x16;
typedef __attribute__((ext_vector_type(8)))  float  floatx8;

__device__ __forceinline__ floatx8 fzero8() {
  floatx8 z;
#pragma unroll
  for (int i = 0; i < 8; ++i) z[i] = 0.0f;
  return z;
}

__device__ __forceinline__ unsigned pack_bf2(float a, float b) {
  union { __bf16 h[2]; unsigned u; } u2;
  u2.h[0] = (__bf16)a; u2.h[1] = (__bf16)b;
  return u2.u;
}

// Load one 16x32 bf16 operand tile (A-matrix layout; also used for B since a
// B "column" is a row of B^T and our B operands are stored transposed).
// Per CDNA5 ISA 16-bit A layout: lane m = lane&15, khalf = (lane>>4)*8,
// elements 0..7 -> K = khalf+0..7, elements 8..15 -> K = khalf+16..23.
__device__ __forceinline__ bf16x16 load_tile(const __bf16* __restrict__ tile,
                                             int ld, int lane) {
  const __bf16* p = tile + (size_t)(lane & 15) * ld + ((lane >> 4) << 3);
  union { bf16x16 v; uint4 q[2]; } u;
  u.q[0] = *reinterpret_cast<const uint4*>(p);
  u.q[1] = *reinterpret_cast<const uint4*>(p + 16);
  return u.v;
}

__device__ __forceinline__ floatx8 wmma_bf16(bf16x16 a, bf16x16 b, floatx8 c) {
  return __builtin_amdgcn_wmma_f32_16x16x32_bf16(false, a, false, b,
                                                 (short)0, c, false, false);
}

__device__ __forceinline__ float sigmoidf_(float x) {
  return 1.0f / (1.0f + __expf(-x));
}

// ---------------------------------------------------------------------------
// K1: f32 -> bf16 conversions
// ---------------------------------------------------------------------------
__global__ void cvt_inputs_kernel(const float* __restrict__ h_s,
                                  const float* __restrict__ W1,
                                  const float* __restrict__ W2,
                                  __bf16* __restrict__ hb,
                                  __bf16* __restrict__ w1b,
                                  __bf16* __restrict__ w2b) {
  const int gtid = blockIdx.x * blockDim.x + threadIdx.x;
  const int stride = gridDim.x * blockDim.x;
  const int nH = kB * kL * kD;
  for (int i = gtid; i < nH; i += stride) hb[i] = (__bf16)h_s[i];
  const int nW = kD * kD;
  for (int i = gtid; i < nW; i += stride) {
    w1b[i] = (__bf16)W1[i];
    w2b[i] = (__bf16)W2[i];
  }
}

// ---------------------------------------------------------------------------
// K2: projections. Block = 256 thr (8 waves), tile 128 rows x 64 cols.
// Each wave: 16 rows x 64 cols, 4 n-tiles x {W1,W2} = 8 accumulators.
// ---------------------------------------------------------------------------
__global__ __launch_bounds__(256) void proj_kernel(
    const __bf16* __restrict__ hb, const __bf16* __restrict__ w1b,
    const __bf16* __restrict__ w2b, const float* __restrict__ m_s,
    const float* __restrict__ w3, __bf16* __restrict__ s1w,
    __bf16* __restrict__ s1T, __bf16* __restrict__ s2b,
    float* __restrict__ s2f) {
  __shared__ float lds1[8][16][64];
  __shared__ float lds2[8][16][64];

  const int tid = threadIdx.x;
  const int wave = tid >> 5;
  const int lane = tid & 31;
  const int hi = lane >> 4;
  const int ln = lane & 15;
  const int n0 = blockIdx.x * 64;            // output-feature tile
  const int row0 = blockIdx.y * 128 + wave * 16;  // global row = b*L + l

  floatx8 acc1[4], acc2[4];
#pragma unroll
  for (int t = 0; t < 4; ++t) { acc1[t] = fzero8(); acc2[t] = fzero8(); }

  for (int k0 = 0; k0 < kD; k0 += 32) {
    bf16x16 a = load_tile(hb + (size_t)row0 * kD + k0, kD, lane);
#pragma unroll
    for (int t = 0; t < 4; ++t) {
      bf16x16 b1 = load_tile(w1b + (size_t)(n0 + 16 * t) * kD + k0, kD, lane);
      acc1[t] = wmma_bf16(a, b1, acc1[t]);
      bf16x16 b2 = load_tile(w2b + (size_t)(n0 + 16 * t) * kD + k0, kD, lane);
      acc2[t] = wmma_bf16(a, b2, acc2[t]);
    }
  }

  // sigmoid + row mask, stage through LDS for coalesced multi-layout stores
  float mrow[8];
#pragma unroll
  for (int r = 0; r < 8; ++r) mrow[r] = m_s[row0 + r + 8 * hi];
#pragma unroll
  for (int t = 0; t < 4; ++t) {
#pragma unroll
    for (int r = 0; r < 8; ++r) {
      lds1[wave][r + 8 * hi][t * 16 + ln] = sigmoidf_(acc1[t][r]) * mrow[r];
      lds2[wave][r + 8 * hi][t * 16 + ln] = sigmoidf_(acc2[t][r]) * mrow[r];
    }
  }
  __syncthreads();

  const int bId = row0 >> 10;       // row0 / L
  const int l0 = row0 & (kL - 1);   // row0 % L
  const int c = 2 * lane;
  const float g0 = w3[n0 + c], g1 = w3[n0 + c + 1];

#pragma unroll 4
  for (int r = 0; r < 16; ++r) {
    const float a0 = lds1[wave][r][c], a1 = lds1[wave][r][c + 1];
    const float b0 = lds2[wave][r][c], b1 = lds2[wave][r][c + 1];
    const size_t base = (size_t)(row0 + r) * kD + n0 + c;
    *reinterpret_cast<unsigned*>(s1w + base) = pack_bf2(a0 * g0, a1 * g1);
    *reinterpret_cast<unsigned*>(s2b + base) = pack_bf2(b0, b1);
    *reinterpret_cast<float2*>(s2f + base) = make_float2(b0, b1);
  }

  // s1 transposed: s1T[b][d][l], 32B per (d) of 16 l-values
#pragma unroll
  for (int q = 0; q < 2; ++q) {
    const int dl = lane + 32 * q;
    unsigned pk[8];
#pragma unroll
    for (int r = 0; r < 8; ++r)
      pk[r] = pack_bf2(lds1[wave][2 * r][dl], lds1[wave][2 * r + 1][dl]);
    __bf16* dst = s1T + (size_t)bId * kD * kL + (size_t)(n0 + dl) * kL + l0;
    reinterpret_cast<uint4*>(dst)[0] = make_uint4(pk[0], pk[1], pk[2], pk[3]);
    reinterpret_cast<uint4*>(dst)[1] = make_uint4(pk[4], pk[5], pk[6], pk[7]);
  }
}

// ---------------------------------------------------------------------------
// K3: scores GEMM + mask + clip + softmax fused. Block: 16 rows x 1024 cols
// (full softmax row must live in one block; 64KB LDS row buffer).
// Wave w covers cols [w*128, w*128+128).
// ---------------------------------------------------------------------------
__global__ __launch_bounds__(256) void scores_softmax_kernel(
    const __bf16* __restrict__ s1w, const __bf16* __restrict__ s2b,
    const float* __restrict__ m_s, float* __restrict__ scf,
    __bf16* __restrict__ pT) {
  __shared__ float sc[16][kL];  // 64 KB

  const int tid = threadIdx.x;
  const int wave = tid >> 5;
  const int lane = tid & 31;
  const int hi = lane >> 4;
  const int ln = lane & 15;
  const int i0 = blockIdx.x * 16;
  const int b = blockIdx.y;
  const int j0 = wave * 128;
  const size_t bLD = (size_t)b * kL * kD;

  floatx8 acc[8];
#pragma unroll
  for (int t = 0; t < 8; ++t) acc[t] = fzero8();

  for (int k0 = 0; k0 < kD; k0 += 32) {
    bf16x16 a = load_tile(s1w + bLD + (size_t)i0 * kD + k0, kD, lane);
#pragma unroll
    for (int t = 0; t < 8; ++t) {
      bf16x16 bm =
          load_tile(s2b + bLD + (size_t)(j0 + 16 * t) * kD + k0, kD, lane);
      acc[t] = wmma_bf16(a, bm, acc[t]);
    }
  }

  float mi[8];
#pragma unroll
  for (int r = 0; r < 8; ++r) mi[r] = m_s[b * kL + i0 + r + 8 * hi];
#pragma unroll
  for (int t = 0; t < 8; ++t) {
    const float mj = m_s[b * kL + j0 + 16 * t + ln];
#pragma unroll
    for (int r = 0; r < 8; ++r) {
      const float mm = mi[r] * mj;
      float v = acc[t][r] * mm - 1.0e11f * (1.0f - mm);
      v = fmaxf(v, -1.0e10f);
      sc[r + 8 * hi][j0 + 16 * t + ln] = v;
    }
  }
  __syncthreads();

  // softmax: each wave owns rows 2*wave, 2*wave+1
#pragma unroll
  for (int rr = 0; rr < 2; ++rr) {
    const int ri = wave * 2 + rr;
    float mx = -3.0e38f;
    for (int j = lane; j < kL; j += 32) mx = fmaxf(mx, sc[ri][j]);
#pragma unroll
    for (int o = 16; o > 0; o >>= 1) mx = fmaxf(mx, __shfl_xor(mx, o, 32));
    float sum = 0.0f;
    for (int j = lane; j < kL; j += 32) {
      const float e = __expf(sc[ri][j] - mx);
      sc[ri][j] = e;
      sum += e;
    }
#pragma unroll
    for (int o = 16; o > 0; o >>= 1) sum += __shfl_xor(sum, o, 32);
    const float inv = 1.0f / sum;
    float* dst = scf + (size_t)b * kL * kL + (size_t)(i0 + ri) * kL;
    for (int j = lane; j < kL; j += 32) {
      const float p = sc[ri][j] * inv;
      sc[ri][j] = p;
      dst[j] = p;  // f32 probs (loss lookups)
    }
  }
  __syncthreads();

  // transposed bf16 probs: pT[b][j][i0..i0+15] (32B per j)
#pragma unroll
  for (int q = 0; q < 4; ++q) {
    const int j = tid + 256 * q;
    unsigned pk[8];
#pragma unroll
    for (int r = 0; r < 8; ++r)
      pk[r] = pack_bf2(sc[2 * r][j], sc[2 * r + 1][j]);
    __bf16* dst = pT + (size_t)b * kL * kL + (size_t)j * kL + i0;
    reinterpret_cast<uint4*>(dst)[0] = make_uint4(pk[0], pk[1], pk[2], pk[3]);
    reinterpret_cast<uint4*>(dst)[1] = make_uint4(pk[4], pk[5], pk[6], pk[7]);
  }
}

// ---------------------------------------------------------------------------
// K4: attn_ctx[b,j,d] = s2f[b,j,d] * sum_i P[b,i,j]*s1[b,i,d]
// A = pT [j,i] (i-contig), B-cols = rows of s1T [d,l] (l-contig).
// Wave = 32 j x 64 d: 2 A tiles x 4 B tiles -> 8 WMMAs per 12 b128 loads
// per k-step (every B tile reused by two WMMAs). Block = 8 waves = 256 j.
// ---------------------------------------------------------------------------
__global__ __launch_bounds__(256) void attn_kernel(
    const __bf16* __restrict__ pT, const __bf16* __restrict__ s1T,
    const float* __restrict__ s2f, float* __restrict__ out) {
  const int tid = threadIdx.x;
  const int wave = tid >> 5;
  const int lane = tid & 31;
  const int hi = lane >> 4;
  const int ln = lane & 15;
  const int d0 = blockIdx.x * 64;
  const int j0 = blockIdx.y * 256 + wave * 32;
  const int b = blockIdx.z;
  const size_t pBase = (size_t)b * kL * kL;
  const size_t tBase = (size_t)b * kD * kL;

  floatx8 acc[2][4];
#pragma unroll
  for (int m = 0; m < 2; ++m)
#pragma unroll
    for (int t = 0; t < 4; ++t) acc[m][t] = fzero8();

  const __bf16* aRow0 = pT + pBase + (size_t)j0 * kL;
  const __bf16* aRow1 = pT + pBase + (size_t)(j0 + 16) * kL;

  for (int k0 = 0; k0 < kL; k0 += 32) {
    bf16x16 a0 = load_tile(aRow0 + k0, kL, lane);
    bf16x16 a1 = load_tile(aRow1 + k0, kL, lane);
    if (k0 + 32 < kL) {  // hint next A k-slice (global_prefetch_b8)
      __builtin_prefetch(aRow0 + (size_t)(lane & 15) * kL + k0 + 32, 0, 1);
      __builtin_prefetch(aRow1 + (size_t)(lane & 15) * kL + k0 + 32, 0, 1);
    }
#pragma unroll
    for (int t = 0; t < 4; ++t) {
      bf16x16 bm =
          load_tile(s1T + tBase + (size_t)(d0 + 16 * t) * kL + k0, kL, lane);
      acc[0][t] = wmma_bf16(a0, bm, acc[0][t]);
      acc[1][t] = wmma_bf16(a1, bm, acc[1][t]);
    }
  }

#pragma unroll
  for (int m = 0; m < 2; ++m) {
#pragma unroll
    for (int t = 0; t < 4; ++t) {
#pragma unroll
      for (int r = 0; r < 8; ++r) {
        const int j = j0 + 16 * m + r + 8 * hi;
        const int d = d0 + 16 * t + ln;
        const size_t idx = (size_t)b * kL * kD + (size_t)j * kD + d;
        out[idx] = s2f[idx] * acc[m][t][r];
      }
    }
  }
}

// ---------------------------------------------------------------------------
// K5: ragged pairwise loss. Single block, deterministic tree reduction.
// ---------------------------------------------------------------------------
__global__ __launch_bounds__(256) void loss_kernel(
    const int* __restrict__ index, const int* __restrict__ index_lens,
    const float* __restrict__ scf, float* __restrict__ loss_out) {
  __shared__ float red[256];
  const int tid = threadIdx.x;
  const float kInvLn10 = 0.43429448190325176f;  // 1/ln(10)
  float acc = 0.0f;
  for (int b = 0; b < kB; ++b) {
    const int n = index_lens[b];
    const float* sb = scf + (size_t)b * kL * kL;
    for (int p = tid; p < n; p += 256) {
      const int a = index[(b * kP + p) * 2 + 0];
      const int c = index[(b * kP + p) * 2 + 1];
      const float sab = sb[(size_t)a * kL + c];
      const float sba = sb[(size_t)c * kL + a];
      const float l1 = fabsf(sab - sba);
      const float l2 = fabsf(-0.05f * __logf(sab + sba) * kInvLn10);
      acc += l1 + l2;
    }
  }
  red[tid] = acc;
  __syncthreads();
  for (int s = 128; s > 0; s >>= 1) {
    if (tid < s) red[tid] += red[tid + s];
    __syncthreads();
  }
  if (tid == 0) *loss_out = red[0];
}

// ---------------------------------------------------------------------------
// Launch
// ---------------------------------------------------------------------------
extern "C" void kernel_launch(void* const* d_in, const int* in_sizes, int n_in,
                              void* d_out, int out_size, void* d_ws,
                              size_t ws_size, hipStream_t stream) {
  (void)in_sizes; (void)n_in; (void)out_size; (void)ws_size;

  const float* h_s = (const float*)d_in[0];
  const float* m_s = (const float*)d_in[1];
  const int* index = (const int*)d_in[2];
  const int* index_lens = (const int*)d_in[3];
  const float* W1 = (const float*)d_in[4];
  const float* W2 = (const float*)d_in[5];
  const float* w3 = (const float*)d_in[6];
  float* out = (float*)d_out;

  char* ws = (char*)d_ws;
  const size_t nBLD = (size_t)kB * kL * kD;  // 4,194,304
  const size_t nBLL = (size_t)kB * kL * kL;  // 16,777,216
  size_t o = 0;
  __bf16* hb  = (__bf16*)(ws + o); o += nBLD * 2;          //  8 MB
  __bf16* s1w = (__bf16*)(ws + o); o += nBLD * 2;          //  8 MB
  __bf16* s1T = (__bf16*)(ws + o); o += nBLD * 2;          //  8 MB
  __bf16* s2b = (__bf16*)(ws + o); o += nBLD * 2;          //  8 MB
  float*  s2f = (float*) (ws + o); o += nBLD * 4;          // 16 MB
  __bf16* pT  = (__bf16*)(ws + o); o += nBLL * 2;          // 32 MB
  float*  scf = (float*) (ws + o); o += nBLL * 4;          // 64 MB
  __bf16* w1b = (__bf16*)(ws + o); o += (size_t)kD * kD * 2;
  __bf16* w2b = (__bf16*)(ws + o); o += (size_t)kD * kD * 2;

  cvt_inputs_kernel<<<2048, 256, 0, stream>>>(h_s, W1, W2, hb, w1b, w2b);

  proj_kernel<<<dim3(kD / 64, (kB * kL) / 128), 256, 0, stream>>>(
      hb, w1b, w2b, m_s, w3, s1w, s1T, s2b, s2f);

  scores_softmax_kernel<<<dim3(kL / 16, kB), 256, 0, stream>>>(
      s1w, s2b, m_s, scf, pT);

  attn_kernel<<<dim3(kD / 64, kL / 256, kB), 256, 0, stream>>>(
      pT, s1T, s2f, out);

  loss_kernel<<<1, 256, 0, stream>>>(index, index_lens, scf, out + nBLD);
}
